// EGNNFlowMatching_54039278518699
// MI455X (gfx1250) — matrix operand order repaired and checked by reading, not compiled, not run on hardware
//
#include <hip/hip_runtime.h>
#include <hip/hip_bf16.h>
#include <math.h>

// ---------------------------------------------------------------------------
// EGNN flow-matching forward for MI455X (gfx1250, wave32, WMMA bf16).
// All GEMMs run as v_wmma_f32_16x16x32_bf16 with f32 accumulation.
// h-row staging uses CDNA5 async global->LDS DMA (ASYNCcnt) via inline asm.
// ---------------------------------------------------------------------------

typedef __attribute__((ext_vector_type(16))) __bf16 v16bf;
typedef __attribute__((ext_vector_type(8)))  float  v8f;

#define NNODES 20000
#define NEDGES 320000
#define HID    128
#define NGAUSS 50
#define NLAYER 4
#define KEPAD  320   // 2*HID + NGAUSS = 306, padded to 10 k-tiles of 32
#define KT_E0  10
#define KT_128 4
#define KT_256 8

__device__ __forceinline__ float silu_f(float x) { return x / (1.0f + __expf(-x)); }

__device__ __forceinline__ v8f wmma_bf16(v16bf a, v16bf b, v8f c) {
  return __builtin_amdgcn_wmma_f32_16x16x32_bf16(false, a, false, b, (short)0, c, false, false);
}

// Async DMA of 16 bytes: global (per-lane 64-bit addr) -> LDS (per-lane 32-bit
// byte offset; low 32 bits of a generic LDS pointer per the aperture rules).
// Tracked on ASYNCcnt; must s_wait_asynccnt 0 before the data is consumed.
__device__ __forceinline__ void async_copy16(unsigned lds_off, unsigned long long gaddr) {
  asm volatile("global_load_async_to_lds_b128 %0, %1, off"
               :: "v"(lds_off), "v"(gaddr)
               : "memory");
}
__device__ __forceinline__ void async_wait0() {
  asm volatile("s_wait_asynccnt 0x0" ::: "memory");
}

// A fragment from LDS, row-major bf16, per ISA 16-bit A layout:
// lane L (row M=L%16): halves 0..7 -> K = kb..kb+7, halves 8..15 -> K = kb+16..kb+23,
// kb = k0 + 8*(L/16). Both runs are 16B-contiguous -> ds_load_b128.
__device__ __forceinline__ v16bf load_a_frag(const __bf16* base, int rowStride, int k0, int lane) {
  const int m  = lane & 15;
  const int kb = k0 + ((lane >> 4) << 3);
  const __bf16* p = base + m * rowStride + kb;
  v16bf a;
#pragma unroll
  for (int j = 0; j < 8; ++j) a[j] = p[j];
#pragma unroll
  for (int j = 0; j < 8; ++j) a[j + 8] = p[16 + j];
  return a;
}

// B fragment from pre-packed global weights: element ((nt*KT+kt)*32+lane)*16+j,
// 32 contiguous bytes per lane -> global_load_b128 x2 (L2-resident).
__device__ __forceinline__ v16bf load_b_frag(const __bf16* packed, int KT, int kt, int nt, int lane) {
  const __bf16* p = packed + ((((size_t)nt * KT + kt) * 32 + lane) << 4);
  v16bf b;
#pragma unroll
  for (int j = 0; j < 16; ++j) b[j] = p[j];
  return b;
}

// ---------------------------------------------------------------------------
// Weight repack: f32 (K x 128 row-major) -> bf16 B-fragment layout.
// B-layout: lane L holds col N=L%16; half j holds K=(L/16)*8+(j&7)+(j>>3)*16 within a k-tile.
// ---------------------------------------------------------------------------
__global__ __launch_bounds__(256) void k_pack_b(const float* __restrict__ W,
                                                int K, int Ncols, int KT,
                                                void* __restrict__ dst_) {
  __bf16* dst = (__bf16*)dst_;
  size_t idx = (size_t)blockIdx.x * 256 + threadIdx.x;   // grid sized exactly KT*4096
  int j    = (int)(idx & 15);
  int lane = (int)((idx >> 4) & 31);
  size_t tile = idx >> 9;
  int kt = (int)(tile % KT);
  int nt = (int)(tile / KT);
  int k = kt * 32 + ((lane >> 4) << 3) + (j & 7) + ((j >> 3) << 4);
  int n = nt * 16 + (lane & 15);
  float v = (k < K && n < Ncols) ? W[(size_t)k * Ncols + n] : 0.0f;
  dst[idx] = (__bf16)v;
}

__global__ __launch_bounds__(256) void k_zero(float* __restrict__ p, int n) {
  int i = blockIdx.x * 256 + threadIdx.x;
  if (i < n) p[i] = 0.0f;
}

// ---------------------------------------------------------------------------
// Per-batch time + cond embeddings (B=16, tiny): tb[b][128]
// ---------------------------------------------------------------------------
__global__ __launch_bounds__(128) void k_batch_embed(
    const float* __restrict__ t, const float* __restrict__ y,
    const float* __restrict__ tW0, const float* __restrict__ tB0,
    const float* __restrict__ tW1, const float* __restrict__ tB1,
    const float* __restrict__ cW0, const float* __restrict__ cB0,
    const float* __restrict__ cW1, const float* __restrict__ cB1,
    float* __restrict__ tb) {
  __shared__ float ht[128], hc[128];
  const int b = blockIdx.x, j = threadIdx.x;
  const float tv = t[b];
  const float yv = (y[b] - 1.0f) * (1.0f / 3.0f);
  float at = tB0[j], ac = cB0[j];
  for (int d = 0; d < 64; ++d) {
    int i = d & 31;
    float f  = __expf(-9.2103403720f * (float)i / 32.0f);  // ln(1e4)
    float et = (d < 32) ? __sinf(tv * f) : __cosf(tv * f);
    float ec = (d < 32) ? __sinf(yv * f) : __cosf(yv * f);
    at += et * tW0[d * 128 + j];
    ac += ec * cW0[d * 128 + j];
  }
  ht[j] = silu_f(at);
  hc[j] = silu_f(ac);
  __syncthreads();
  float o = tB1[j] + cB1[j];
  for (int k = 0; k < 128; ++k) o += ht[k] * tW1[k * 128 + j] + hc[k] * cW1[k * 128 + j];
  tb[b * 128 + j] = o;
}

// ---------------------------------------------------------------------------
// Node embedding: h = MLP(pos) + tb[batch]  (K=3 -> scalar is fine, <1% FLOPs)
// Also copies pos into mutable pos_work. Writes h as bf16.
// ---------------------------------------------------------------------------
__global__ __launch_bounds__(128) void k_node_embed(
    const float* __restrict__ pos, const int* __restrict__ batch,
    const float* __restrict__ W0, const float* __restrict__ B0,
    const float* __restrict__ W1, const float* __restrict__ B1,
    const float* __restrict__ tb,
    void* __restrict__ hbf_, float* __restrict__ posw) {
  __bf16* hbf = (__bf16*)hbf_;
  __shared__ float hid[128];
  const int i = blockIdx.x, j = threadIdx.x;
  const float p0 = pos[i * 3], p1 = pos[i * 3 + 1], p2 = pos[i * 3 + 2];
  hid[j] = silu_f(B0[j] + p0 * W0[j] + p1 * W0[128 + j] + p2 * W0[256 + j]);
  if (j < 3) posw[i * 3 + j] = pos[i * 3 + j];
  __syncthreads();
  float o = B1[j] + tb[batch[i] * 128 + j];
  for (int k = 0; k < 128; ++k) o += hid[k] * W1[k * 128 + j];
  hbf[(size_t)i * HID + j] = (__bf16)o;
}

// ---------------------------------------------------------------------------
// Edge kernel: one 16-edge tile per wave, 4 waves/block (all tiles exact).
//   A = [h[src] | h[dst] | gaussian_smear | 0pad]  (16 x 320 bf16 in LDS,
//       h halves staged via global_load_async_to_lds_b128)
//   hidden   = SiLU(A @ W0 + b0)            : 10 WMMA / n-tile
//   ef       = hidden @ W1 + b1             : 4  WMMA / n-tile  -> atomicAdd agg[dst]
//   ch       = SiLU(ef @ Wc0 + bc0)         : 4  WMMA / n-tile
//   coordw   = ch . wc1 + bc1               -> atomicAdd dpos[dst]
// ---------------------------------------------------------------------------
__global__ __launch_bounds__(128) void k_edge(
    const float* __restrict__ posw, const void* __restrict__ hbf_,
    const int* __restrict__ esrc, const int* __restrict__ edst,
    const void* __restrict__ W0p_, const float* __restrict__ b0,
    const void* __restrict__ W1p_, const float* __restrict__ b1,
    const void* __restrict__ Wc0p_, const float* __restrict__ bc0,
    const float* __restrict__ wc1, const float* __restrict__ bc1,
    float* __restrict__ agg, float* __restrict__ dpos) {
  const __bf16* hbf  = (const __bf16*)hbf_;
  const __bf16* W0p  = (const __bf16*)W0p_;
  const __bf16* W1p  = (const __bf16*)W1p_;
  const __bf16* Wc0p = (const __bf16*)Wc0p_;

  __shared__ __bf16 sA[4][16][KEPAD];   // 40960 B : edge_in, later reused for edge_feat
  __shared__ __bf16 sH[4][16][HID];     // 16384 B : hidden
  __shared__ float  sVec[4][16][4];     // evec + dist
  __shared__ float  sCw[4][16];         // coord weight accumulators
  __shared__ int    sIdx[4][16][2];     // src / dst

  const int w = threadIdx.x >> 5, lane = threadIdx.x & 31;
  const int tile = blockIdx.x * 4 + w;
  const int e0 = tile * 16;
  const int lo = lane & 15, mbase = (lane >> 4) << 3;

  // warm packed-weight streams (global_prefetch_b8; L2 hints)
  __builtin_prefetch(W0p, 0, 2);
  __builtin_prefetch(W1p, 0, 2);
  __builtin_prefetch(Wc0p, 0, 2);

  if (lane < 16) {
    const int e = e0 + lane;
    const int s = esrc[e], d = edst[e];
    sIdx[w][lane][0] = s; sIdx[w][lane][1] = d;
    const float dx = posw[d * 3 + 0] - posw[s * 3 + 0];
    const float dy = posw[d * 3 + 1] - posw[s * 3 + 1];
    const float dz = posw[d * 3 + 2] - posw[s * 3 + 2];
    const float dist = sqrtf(dx * dx + dy * dy + dz * dz);
    sVec[w][lane][0] = dx; sVec[w][lane][1] = dy;
    sVec[w][lane][2] = dz; sVec[w][lane][3] = dist;
    sCw[w][lane] = bc1[0];
  }
  __syncthreads();

  // Stage A rows. h halves via async DMA (memory->LDS, no VGPR round-trip):
  // lanes 0..15 carry the src half, lanes 16..31 the dst half, 16B per lane.
  {
    const int half = lane >> 4;
    const int sub  = lane & 15;
    for (int r = 0; r < 16; ++r) {
      const int node = sIdx[w][r][half];
      const unsigned long long ga =
          (unsigned long long)(size_t)(hbf + (size_t)node * HID) + (unsigned long long)(sub * 16);
      const unsigned la =
          (unsigned)(size_t)(&sA[w][r][half * 128]) + (unsigned)(sub * 16);
      async_copy16(la, ga);
    }
    // Gaussian smear features (normal ds stores, cols 256..319)
    for (int r = 0; r < 16; ++r) {
      const float dist = sVec[w][r][3];
#pragma unroll
      for (int g = lane; g < 64; g += 32) {
        float v = 0.0f;
        if (g < NGAUSS) {
          const float dd = dist - (float)g * (5.0f / 49.0f);
          v = __expf(-48.0200004f * dd * dd);        // -0.5/(5/49)^2
        }
        sA[w][r][256 + g] = (__bf16)v;
      }
    }
    async_wait0();
  }
  __syncthreads();

  // GEMM1: hidden = SiLU(A @ W0 + b0)
  for (int nt = 0; nt < 8; ++nt) {
    v8f acc = {};
#pragma unroll
    for (int kt = 0; kt < KT_E0; ++kt) {
      v16bf a = load_a_frag(&sA[w][0][0], KEPAD, kt * 32, lane);
      v16bf b = load_b_frag(W0p, KT_E0, kt, nt, lane);
      acc = wmma_bf16(a, b, acc);
    }
    const int n = nt * 16 + lo;
    const float bias = b0[n];
#pragma unroll
    for (int r = 0; r < 8; ++r) sH[w][mbase + r][n] = (__bf16)silu_f(acc[r] + bias);
  }
  __syncthreads();

  // GEMM2: edge_feat = hidden @ W1 + b1 ; scatter-add into agg[dst]
  for (int nt = 0; nt < 8; ++nt) {
    v8f acc = {};
#pragma unroll
    for (int kt = 0; kt < KT_128; ++kt) {
      v16bf a = load_a_frag(&sH[w][0][0], HID, kt * 32, lane);
      v16bf b = load_b_frag(W1p, KT_128, kt, nt, lane);
      acc = wmma_bf16(a, b, acc);
    }
    const int n = nt * 16 + lo;
    const float bias = b1[n];
#pragma unroll
    for (int r = 0; r < 8; ++r) {
      const float ef = acc[r] + bias;
      const int m = mbase + r;
      atomicAdd(&agg[(size_t)sIdx[w][m][1] * HID + n], ef);
      sA[w][m][n] = (__bf16)ef;                 // reuse sA as edge_feat store
    }
  }
  __syncthreads();

  // GEMM3: coord hidden = SiLU(ef @ Wc0 + bc0); row-reduce with wc1
  float partial[8] = {0, 0, 0, 0, 0, 0, 0, 0};
  for (int nt = 0; nt < 8; ++nt) {
    v8f acc = {};
#pragma unroll
    for (int kt = 0; kt < KT_128; ++kt) {
      v16bf a = load_a_frag(&sA[w][0][0], KEPAD, kt * 32, lane);
      v16bf b = load_b_frag(Wc0p, KT_128, kt, nt, lane);
      acc = wmma_bf16(a, b, acc);
    }
    const int n = nt * 16 + lo;
    const float bias = bc0[n], wn = wc1[n];
#pragma unroll
    for (int r = 0; r < 8; ++r) partial[r] += silu_f(acc[r] + bias) * wn;
  }
#pragma unroll
  for (int r = 0; r < 8; ++r) atomicAdd(&sCw[w][mbase + r], partial[r]);  // ds_add_f32
  __syncthreads();

  if (lane < 16) {
    const int m = lane;
    const float inv = sCw[w][m] / (sVec[w][m][3] + 1e-8f);
    const int d = sIdx[w][m][1];
    atomicAdd(&dpos[d * 3 + 0], sVec[w][m][0] * inv);
    atomicAdd(&dpos[d * 3 + 1], sVec[w][m][1] * inv);
    atomicAdd(&dpos[d * 3 + 2], sVec[w][m][2] * inv);
  }
}

// ---------------------------------------------------------------------------
// Node kernel: xn=[h|agg] (K=256); h = LN(xn@W0->SiLU->@W1 + xn@Ws); pos += dpos
// One 16-node tile per wave, 2 waves/block. h rows staged via async DMA.
// ---------------------------------------------------------------------------
__global__ __launch_bounds__(64) void k_node(
    void* __restrict__ hbf_, const float* __restrict__ agg,
    const void* __restrict__ W0p_, const float* __restrict__ b0,
    const void* __restrict__ W1p_, const float* __restrict__ b1,
    const void* __restrict__ Wsp_, const float* __restrict__ bs,
    const float* __restrict__ lng, const float* __restrict__ lnb,
    float* __restrict__ posw, const float* __restrict__ dpos) {
  __bf16* hbf = (__bf16*)hbf_;
  const __bf16* W0p = (const __bf16*)W0p_;
  const __bf16* W1p = (const __bf16*)W1p_;
  const __bf16* Wsp = (const __bf16*)Wsp_;

  __shared__ __bf16 nA[2][16][256];
  __shared__ __bf16 nH[2][16][HID];
  __shared__ float  nS[2][16][HID];

  const int w = threadIdx.x >> 5, lane = threadIdx.x & 31;
  const int tile = blockIdx.x * 2 + w;
  const int i0 = tile * 16;
  const int lo = lane & 15, mbase = (lane >> 4) << 3;

  __builtin_prefetch(W0p, 0, 2);
  __builtin_prefetch(W1p, 0, 2);
  __builtin_prefetch(Wsp, 0, 2);

  // h rows: two rows per async step (32 lanes x 16B)
  {
    const int rsel = lane >> 4, sub = lane & 15;
    for (int r = 0; r < 16; r += 2) {
      const int rr = r + rsel;
      const unsigned long long ga =
          (unsigned long long)(size_t)(hbf + (size_t)(i0 + rr) * HID) + (unsigned long long)(sub * 16);
      const unsigned la = (unsigned)(size_t)(&nA[w][rr][0]) + (unsigned)(sub * 16);
      async_copy16(la, ga);
    }
  }
  // agg half (f32 -> bf16 conversion, normal path)
  for (int r = 0; r < 16; ++r) {
    const size_t i = (size_t)(i0 + r);
#pragma unroll
    for (int c = lane; c < HID; c += 32) nA[w][r][128 + c] = (__bf16)agg[i * HID + c];
  }
  async_wait0();
  __syncthreads();

  for (int nt = 0; nt < 8; ++nt) {
    v8f acc = {};
#pragma unroll
    for (int kt = 0; kt < KT_256; ++kt) {
      v16bf a = load_a_frag(&nA[w][0][0], 256, kt * 32, lane);
      v16bf b = load_b_frag(W0p, KT_256, kt, nt, lane);
      acc = wmma_bf16(a, b, acc);
    }
    const int n = nt * 16 + lo;
    const float bias = b0[n];
#pragma unroll
    for (int r = 0; r < 8; ++r) nH[w][mbase + r][n] = (__bf16)silu_f(acc[r] + bias);
  }
  __syncthreads();

  for (int nt = 0; nt < 8; ++nt) {
    v8f acc = {}, accS = {};
#pragma unroll
    for (int kt = 0; kt < KT_128; ++kt) {
      v16bf a = load_a_frag(&nH[w][0][0], HID, kt * 32, lane);
      v16bf b = load_b_frag(W1p, KT_128, kt, nt, lane);
      acc = wmma_bf16(a, b, acc);
    }
#pragma unroll
    for (int kt = 0; kt < KT_256; ++kt) {
      v16bf a = load_a_frag(&nA[w][0][0], 256, kt * 32, lane);
      v16bf b = load_b_frag(Wsp, KT_256, kt, nt, lane);
      accS = wmma_bf16(a, b, accS);
    }
    const int n = nt * 16 + lo;
    const float bsum = b1[n] + bs[n];
#pragma unroll
    for (int r = 0; r < 8; ++r) nS[w][mbase + r][n] = acc[r] + accS[r] + bsum;
  }
  __syncthreads();

  if (lane < 16) {
    const int r = lane;
    const size_t i = (size_t)(i0 + r);
    float mu = 0.0f;
    for (int c = 0; c < HID; ++c) mu += nS[w][r][c];
    mu *= (1.0f / HID);
    float var = 0.0f;
    for (int c = 0; c < HID; ++c) { const float d = nS[w][r][c] - mu; var += d * d; }
    var *= (1.0f / HID);
    const float rs = rsqrtf(var + 1e-5f);
    for (int c = 0; c < HID; ++c)
      hbf[i * HID + c] = (__bf16)((nS[w][r][c] - mu) * rs * lng[c] + lnb[c]);
    posw[i * 3 + 0] += dpos[i * 3 + 0];
    posw[i * 3 + 1] += dpos[i * 3 + 1];
    posw[i * 3 + 2] += dpos[i * 3 + 2];
  }
}

// ---------------------------------------------------------------------------
// Output head: out = SiLU(h @ Wo0 + bo0) @ Wo1 + bo1   (N x 3, f32)
// ---------------------------------------------------------------------------
__global__ __launch_bounds__(64) void k_out(
    const void* __restrict__ hbf_,
    const void* __restrict__ W0p_, const float* __restrict__ b0,
    const float* __restrict__ W1, const float* __restrict__ b1,
    float* __restrict__ out) {
  const __bf16* hbf = (const __bf16*)hbf_;
  const __bf16* W0p = (const __bf16*)W0p_;
  __shared__ __bf16 oA[2][16][HID];
  __shared__ float  oT[2][16][HID];
  const int w = threadIdx.x >> 5, lane = threadIdx.x & 31;
  const int tile = blockIdx.x * 2 + w;
  const int i0 = tile * 16;
  const int lo = lane & 15, mbase = (lane >> 4) << 3;

  __builtin_prefetch(W0p, 0, 2);

  {
    const int rsel = lane >> 4, sub = lane & 15;
    for (int r = 0; r < 16; r += 2) {
      const int rr = r + rsel;
      const unsigned long long ga =
          (unsigned long long)(size_t)(hbf + (size_t)(i0 + rr) * HID) + (unsigned long long)(sub * 16);
      const unsigned la = (unsigned)(size_t)(&oA[w][rr][0]) + (unsigned)(sub * 16);
      async_copy16(la, ga);
    }
    async_wait0();
  }
  __syncthreads();

  for (int nt = 0; nt < 8; ++nt) {
    v8f acc = {};
#pragma unroll
    for (int kt = 0; kt < KT_128; ++kt) {
      v16bf a = load_a_frag(&oA[w][0][0], HID, kt * 32, lane);
      v16bf b = load_b_frag(W0p, KT_128, kt, nt, lane);
      acc = wmma_bf16(a, b, acc);
    }
    const int n = nt * 16 + lo;
    const float bias = b0[n];
#pragma unroll
    for (int r = 0; r < 8; ++r) oT[w][mbase + r][n] = silu_f(acc[r] + bias);
  }
  __syncthreads();

  for (int idx = lane; idx < 48; idx += 32) {
    const int r = idx / 3, c = idx - 3 * r;
    float o = b1[c];
    for (int k = 0; k < HID; ++k) o += oT[w][r][k] * W1[k * 3 + c];
    out[(size_t)(i0 + r) * 3 + c] = o;
  }
}

// ---------------------------------------------------------------------------
// Host: input pytree mapping (top-level in setup_inputs() insertion order;
// params flattened with sorted dict keys at each level, lists in order):
//  0 pos  1 t  2 y
//  3..6   cond_mlp   (l0.b, l0.w, l1.b, l1.w)
//  7+16l  layers[l]: coord(l0.b,l0.w,l1.b,l1.w), edge(l0.b,l0.w,l1.b,l1.w),
//                    ln_b, ln_g, node(l0.b,l0.w,l1.b,l1.w), short.b, short.w
//  71..74 node_embed   75..78 out_net   79..82 time_mlp
//  83 edge_index (2 x E)   84 batch
// ---------------------------------------------------------------------------
extern "C" void kernel_launch(void* const* d_in, const int* in_sizes, int n_in,
                              void* d_out, int out_size, void* d_ws, size_t ws_size,
                              hipStream_t stream) {
  if (n_in < 85) return;
  const float* pos = (const float*)d_in[0];
  const float* t   = (const float*)d_in[1];
  const float* y   = (const float*)d_in[2];
  const int* eidx  = (const int*)d_in[83];
  const int* batch = (const int*)d_in[84];
  const int* esrc = eidx;
  const int* edst = eidx + NEDGES;

  auto F = [&](int i) { return (const float*)d_in[i]; };

  // ---- workspace carve-out ----
  char* ws = (char*)d_ws;
  size_t off = 0;
  auto carve = [&](size_t bytes) -> void* {
    void* p = ws + off;
    off += (bytes + 255) & ~(size_t)255;
    return p;
  };
  float* agg  = (float*)carve((size_t)NNODES * HID * 4);
  float* dpos = (float*)carve((size_t)NNODES * 3 * 4);
  float* posw = (float*)carve((size_t)NNODES * 3 * 4);
  void*  hbf  = carve((size_t)NNODES * HID * 2);
  float* tb   = (float*)carve(16 * 128 * 4);
  void *pkE0[NLAYER], *pkE1[NLAYER], *pkN0[NLAYER], *pkNS[NLAYER], *pkN1[NLAYER], *pkC0[NLAYER];
  for (int l = 0; l < NLAYER; ++l) {
    pkE0[l] = carve((size_t)KT_E0  * 4096 * 2);
    pkE1[l] = carve((size_t)KT_128 * 4096 * 2);
    pkN0[l] = carve((size_t)KT_256 * 4096 * 2);
    pkNS[l] = carve((size_t)KT_256 * 4096 * 2);
    pkN1[l] = carve((size_t)KT_128 * 4096 * 2);
    pkC0[l] = carve((size_t)KT_128 * 4096 * 2);
  }
  void* pkO0 = carve((size_t)KT_128 * 4096 * 2);
  if (off > ws_size) return;

  // ---- weight repack (bf16, B-fragment layout) ----
  for (int l = 0; l < NLAYER; ++l) {
    const int base = 7 + l * 16;
    k_pack_b<<<KT_E0  * 16, 256, 0, stream>>>(F(base + 5),  306, 128, KT_E0,  pkE0[l]); // edge l0.w
    k_pack_b<<<KT_128 * 16, 256, 0, stream>>>(F(base + 7),  128, 128, KT_128, pkE1[l]); // edge l1.w
    k_pack_b<<<KT_256 * 16, 256, 0, stream>>>(F(base + 11), 256, 128, KT_256, pkN0[l]); // node l0.w
    k_pack_b<<<KT_256 * 16, 256, 0, stream>>>(F(base + 15), 256, 128, KT_256, pkNS[l]); // short.w
    k_pack_b<<<KT_128 * 16, 256, 0, stream>>>(F(base + 13), 128, 128, KT_128, pkN1[l]); // node l1.w
    k_pack_b<<<KT_128 * 16, 256, 0, stream>>>(F(base + 1),  128, 128, KT_128, pkC0[l]); // coord l0.w
  }
  k_pack_b<<<KT_128 * 16, 256, 0, stream>>>(F(76), 128, 128, KT_128, pkO0);             // out l0.w

  // ---- embeddings ----
  k_batch_embed<<<16, 128, 0, stream>>>(t, y,
      F(80), F(79), F(82), F(81),   // time_mlp: l0.w, l0.b, l1.w, l1.b
      F(4),  F(3),  F(6),  F(5),    // cond_mlp
      tb);
  k_node_embed<<<NNODES, 128, 0, stream>>>(pos, batch,
      F(72), F(71), F(74), F(73),   // node_embed
      tb, hbf, posw);

  // ---- message-passing layers ----
  for (int l = 0; l < NLAYER; ++l) {
    const int base = 7 + l * 16;
    k_zero<<<(NNODES * HID + 255) / 256, 256, 0, stream>>>(agg, NNODES * HID);
    k_zero<<<(NNODES * 3 + 255) / 256, 256, 0, stream>>>(dpos, NNODES * 3);
    k_edge<<<NEDGES / 16 / 4, 128, 0, stream>>>(
        posw, hbf, esrc, edst,
        pkE0[l], F(base + 4),        // edge W0, b0
        pkE1[l], F(base + 6),        // edge W1, b1
        pkC0[l], F(base + 0),        // coord W0, b0
        F(base + 3), F(base + 2),    // coord w1 (128x1), b1 (1)
        agg, dpos);
    k_node<<<NNODES / 16 / 2, 64, 0, stream>>>(
        hbf, agg,
        pkN0[l], F(base + 10),       // node W0, b0
        pkN1[l], F(base + 12),       // node W1, b1
        pkNS[l], F(base + 14),       // short W, b
        F(base + 9), F(base + 8),    // ln_g, ln_b
        posw, dpos);
  }

  // ---- output head ----
  k_out<<<NNODES / 16 / 2, 64, 0, stream>>>(hbf, pkO0, F(75), F(78), F(77),
                                            (float*)d_out);
}
